// Decoder_9122510537029
// MI455X (gfx1250) — matrix operand order repaired and checked by reading, not compiled, not run on hardware
//
#include <hip/hip_runtime.h>

// ---------------------------------------------------------------------------
// Types for CDNA5 WMMA (wave32, 16x16x32 bf16 -> f32) and TDM descriptors
// ---------------------------------------------------------------------------
typedef __bf16 bf16_t;
typedef __attribute__((ext_vector_type(16))) __bf16 bf16x16;
typedef __attribute__((ext_vector_type(8)))  float  f32x8;
typedef __attribute__((ext_vector_type(4)))  unsigned int u32x4;
typedef __attribute__((ext_vector_type(8)))  int i32x8;
typedef __attribute__((ext_vector_type(4)))  int i32x4;

union FragU { float4 q[2]; bf16x16 v; };

__device__ __forceinline__ f32x8 wmma_bf16(bf16x16 a, bf16x16 b, f32x8 c) {
  return __builtin_amdgcn_wmma_f32_16x16x32_bf16(
      /*neg_a=*/false, a, /*neg_b=*/false, b,
      /*c_mod=*/(short)0, c, /*reuse_a=*/false, /*reuse_b=*/false);
}

// A-matrix 16x32 bf16 fragment, contiguous k = 16*(lane/16) + s (two 16B loads per lane).
__device__ __forceinline__ bf16x16 load_frag_A(const bf16_t* A, int ld, int m0, int kc) {
  int lane = threadIdx.x & 31;
  const char* row = (const char*)(A + (size_t)(m0 + (lane & 15)) * (size_t)ld + kc);
  int hb = (lane >> 4) * 32;
  FragU f;
  f.q[0] = *(const float4*)(row + hb);
  f.q[1] = *(const float4*)(row + hb + 16);
  return f.v;
}

// B-matrix 32x16 bf16 fragment from transposed weights WT[N][K] in GLOBAL
// memory (one contiguous 32B read) + prefetch of the next K-chunk.
__device__ __forceinline__ bf16x16 load_frag_B(const bf16_t* WT, int ld, int n0, int kc) {
  int lane = threadIdx.x & 31;
  const char* row = (const char*)(WT + (size_t)(n0 + (lane & 15)) * (size_t)ld + kc);
  int hb = (lane >> 4) * 32;
  __builtin_prefetch(row + 64, 0, 1);   // next K-chunk of this weight row
  FragU f;
  f.q[0] = *(const float4*)(row + hb);
  f.q[1] = *(const float4*)(row + hb + 16);
  return f.v;
}

// Same B-fragment gather but from an LDS-resident tile (ds_load path).
__device__ __forceinline__ bf16x16 load_frag_L(const bf16_t* WL, int ld, int n0, int kc) {
  int lane = threadIdx.x & 31;
  const char* row = (const char*)(WL + (size_t)(n0 + (lane & 15)) * (size_t)ld + kc);
  int hb = (lane >> 4) * 32;
  FragU f;
  f.q[0] = *(const float4*)(row + hb);
  f.q[1] = *(const float4*)(row + hb + 16);
  return f.v;
}

__device__ __forceinline__ float sigmoidf(float x) {
  return 1.0f / (1.0f + __expf(-x));
}

// ---------------------------------------------------------------------------
// Tensor Data Mover: stage a [tile_z][tile_rows][tile_k] bf16 tile from
// global (row stride s0 elems, z stride s1 elems) into contiguous LDS.
// D# layout per CDNA5 ISA ch.8 (group0 128b, group1 256b, group2/3 128b).
// clang-23 builtin: (u32x4 g0, i32x8 g1, i32x4 g2, i32x4 g3, i32x8 g4, i32 cpol)
// ---------------------------------------------------------------------------
__device__ __forceinline__ void tdm_stage(const bf16_t* gsrc, unsigned lds_off,
                                          unsigned tile_k, unsigned tile_rows,
                                          unsigned tile_z,
                                          unsigned long long s0,
                                          unsigned long long s1) {
  unsigned long long ga = (unsigned long long)(uintptr_t)gsrc;
  u32x4 g0;
  g0[0] = 1u;                                        // count=1 valid user D#
  g0[1] = lds_off;                                   // lds_addr (bytes)
  g0[2] = (unsigned)(ga & 0xffffffffu);              // global_addr[31:0]
  g0[3] = (unsigned)((ga >> 32) & 0x01ffffffu) | (2u << 30);   // [56:32]|type=2
  unsigned td0 = tile_k, td1 = tile_rows;
  i32x8 g1;
  g1[0] = (int)(1u << 16);                           // data_size=1 (2 bytes)
  g1[1] = (int)((td0 & 0xffffu) << 16);              // tensor_dim0[15:0]
  g1[2] = (int)(((td0 >> 16) & 0xffffu) | ((td1 & 0xffffu) << 16));
  g1[3] = (int)(((td1 >> 16) & 0xffffu) | ((tile_k & 0xffffu) << 16)); // tile_dim0
  g1[4] = (int)((tile_rows & 0xffffu) | ((tile_z & 0xffffu) << 16));   // tile_dim1,2
  g1[5] = (int)(s0 & 0xffffffffu);                   // tensor_dim0_stride[31:0]
  g1[6] = (int)(((s0 >> 32) & 0xffffu) | ((s1 & 0xffffu) << 16));
  g1[7] = (int)((s1 >> 16) & 0xffffffffu);           // tensor_dim1_stride[47:16]
  i32x4 g2;
  g2[0] = (int)(tile_z ? tile_z : 1);                // tensor_dim2
  g2[1] = 1;                                         // tensor_dim3 (unused)
  g2[2] = 0; g2[3] = 0;                              // dim2_stride / tile_dim3
  i32x4 g3; g3[0] = 0; g3[1] = 0; g3[2] = 0; g3[3] = 0;
  i32x8 g4;
  g4[0] = 0; g4[1] = 0; g4[2] = 0; g4[3] = 0;
  g4[4] = 0; g4[5] = 0; g4[6] = 0; g4[7] = 0;
  __builtin_amdgcn_tensor_load_to_lds(g0, g1, g2, g3, g4, 0);
}

// ---------------------------------------------------------------------------
// Single-source WMMA GEMM: C[M][N] = A[M][K] @ WT^T (WT stored [N][K]).
// Block = 64x64 tile, 8 waves x 2 accumulators. mode 0: f32; 1: relu->bf16.
// ---------------------------------------------------------------------------
__global__ __launch_bounds__(256) void gemm1_kernel(
    const bf16_t* A, const bf16_t* WT, int ldw, int Ktot,
    float* outF, bf16_t* outB, int N, int mode)
{
  int w = threadIdx.x >> 5, lane = threadIdx.x & 31;
  int mt  = blockIdx.y * 64 + (w & 3) * 16;
  int nt0 = blockIdx.x * 64 + (w >> 2) * 32;
  f32x8 acc0 = {}; f32x8 acc1 = {};
  for (int k = 0; k < Ktot; k += 32) {
    bf16x16 a  = load_frag_A(A, Ktot, mt, k);
    bf16x16 b0 = load_frag_B(WT, ldw, nt0, k);
    bf16x16 b1 = load_frag_B(WT, ldw, nt0 + 16, k);
    acc0 = wmma_bf16(a, b0, acc0);
    acc1 = wmma_bf16(a, b1, acc1);
  }
  int half = lane >> 4, nl = lane & 15;
  for (int r = 0; r < 8; ++r) {
    int m = mt + r + half * 8;
    int n0 = nt0 + nl, n1 = nt0 + 16 + nl;
    if (mode == 0) {
      outF[(size_t)m * N + n0] = acc0[r];
      outF[(size_t)m * N + n1] = acc1[r];
    } else {
      outB[(size_t)m * N + n0] = (bf16_t)fmaxf(acc0[r], 0.0f);
      outB[(size_t)m * N + n1] = (bf16_t)fmaxf(acc1[r], 0.0f);
    }
  }
}

// ---------------------------------------------------------------------------
// Fused LSTM step, TDM-staged weights. Block nb owns hidden cols
// [nb*16, nb*16+16): all 4 gate tiles (WT rows g*1024+nb*16+c). Weights are
// DMA'd per 256-wide K-chunk as a 3D tile [4 gates][16 rows][256 k] into a
// double-buffered LDS tile by wave 0 (TENSORcnt + barrier sync), WMMAs read
// B-fragments from LDS, A-fragments stream from global (L2 resident).
// ---------------------------------------------------------------------------
__global__ __launch_bounds__(256) void lstm_kernel(
    const bf16_t* A0, int l0, const bf16_t* A1, int l1, const bf16_t* A2, int l2,
    const bf16_t* WT, int ldw, int Ktot, const float* bias,
    float* cbuf, float* hF, bf16_t* hB)
{
  __shared__ __align__(128) bf16_t sW[2][64][256];   // [buf][gate*16+row][k]
  __shared__ float g[4][64][16];
  int nb = blockIdx.x;                               // hidden-column block
  int w = threadIdx.x >> 5, lane = threadIdx.x & 31;
  int mt = (w & 3) * 16;
  int gate0 = (w >> 2) * 2;
  int NC = Ktot >> 8;                                // 256-wide K-chunks
  bool issuer = (w == 0);                            // wave 0 drives the TDM

  if (issuer)
    tdm_stage(WT + (size_t)nb * 16 * ldw,
              (unsigned)(uintptr_t)&sW[0][0][0],
              256, 16, 4, (unsigned long long)ldw,
              (unsigned long long)ldw * 1024ull);

  f32x8 acc0 = {}; f32x8 acc1 = {};
  for (int c = 0; c < NC; ++c) {
    int kb = c << 8;
    // A-piece selection once per chunk (all pieces are multiples of 256)
    const bf16_t* Asrc; int aLd, aOff;
    if (kb < l0)           { Asrc = A0; aLd = l0; aOff = kb; }
    else if (kb - l0 < l1) { Asrc = A1; aLd = l1; aOff = kb - l0; }
    else                   { Asrc = A2; aLd = l2; aOff = kb - l0 - l1; }

    if (issuer) {
      if (c + 1 < NC) {
        tdm_stage(WT + (size_t)nb * 16 * ldw + ((c + 1) << 8),
                  (unsigned)(uintptr_t)&sW[(c + 1) & 1][0][0],
                  256, 16, 4, (unsigned long long)ldw,
                  (unsigned long long)ldw * 1024ull);
        __builtin_amdgcn_s_wait_tensorcnt(1);        // chunk c landed
      } else {
        __builtin_amdgcn_s_wait_tensorcnt(0);
      }
    }
    __syncthreads();                                 // publish chunk c
    const bf16_t* sbuf = &sW[c & 1][0][0];
    for (int k = 0; k < 256; k += 32) {
      bf16x16 a  = load_frag_A(Asrc, aLd, mt, aOff + k);
      bf16x16 b0 = load_frag_L(sbuf, 256, gate0 * 16, k);
      bf16x16 b1 = load_frag_L(sbuf, 256, gate0 * 16 + 16, k);
      acc0 = wmma_bf16(a, b0, acc0);
      acc1 = wmma_bf16(a, b1, acc1);
    }
    __syncthreads();                                 // done reading buffer
  }

  int half = lane >> 4, nl = lane & 15;
  for (int r = 0; r < 8; ++r) {
    int m = mt + r + half * 8;
    g[gate0    ][m][nl] = acc0[r];
    g[gate0 + 1][m][nl] = acc1[r];
  }
  __syncthreads();
  for (int idx = threadIdx.x; idx < 64 * 16; idx += 256) {
    int m = idx >> 4, c = idx & 15;
    int hcol = nb * 16 + c;
    float gi = g[0][m][c] + bias[hcol];
    float gf = g[1][m][c] + bias[1024 + hcol];
    float gg = g[2][m][c] + bias[2048 + hcol];
    float go = g[3][m][c] + bias[3072 + hcol];
    float cO = cbuf[m * 1024 + hcol];
    float cN = sigmoidf(gf) * cO + sigmoidf(gi) * tanhf(gg);
    float h  = sigmoidf(go) * tanhf(cN);
    cbuf[m * 1024 + hcol] = cN;
    hF[m * 1024 + hcol]   = h;
    hB[m * 1024 + hcol]   = (bf16_t)h;
  }
}

// ---------------------------------------------------------------------------
// Fused location-sensitive attention, one block per batch element.
// ---------------------------------------------------------------------------
__global__ __launch_bounds__(256) void attention_kernel(
    const float* ahF, const float* enc, const float* pm,
    const bf16_t* WqT, const bf16_t* WlocT, const float* Kloc, const float* Wv,
    const int* tlen, float* aw, float* awc, bf16_t* ctxB, float* align_out, int t)
{
  __shared__ float s_aw[512], s_awc[512], s_q[128], s_e[512], s_red[8];
  __shared__ __align__(64) bf16_t s_loc[512][32];
  int b = blockIdx.x, tid = threadIdx.x;
  int w = tid >> 5, lane = tid & 31, half = lane >> 4, nl = lane & 15;

  for (int i = tid; i < 512; i += 256) {
    s_aw[i]  = aw[b * 512 + i];
    s_awc[i] = awc[b * 512 + i];
    s_e[i]   = 0.0f;
  }
  if (tid < 128) {                      // query = ah[b] @ Wq
    const float*  ahr = ahF + (size_t)b * 1024;
    const bf16_t* wr  = WqT + (size_t)tid * 1024;
    float q = 0.0f;
    for (int k = 0; k < 1024; ++k) q += ahr[k] * (float)wr[k];
    s_q[tid] = q;
  }
  __syncthreads();

  for (int tt = tid; tt < 512; tt += 256) {   // location conv, pad 15
    for (int f = 0; f < 32; ++f) {
      const float* kf = Kloc + f * 62;
      float acc = 0.0f;
      for (int k = 0; k < 31; ++k) {
        int p = tt + k - 15;
        if (p >= 0 && p < 512) acc += kf[k] * s_aw[p] + kf[31 + k] * s_awc[p];
      }
      s_loc[tt][f] = (bf16_t)acc;
    }
  }
  __syncthreads();

  // energies via WMMA: [512x32]@[32x128], tanh(.+q+pm).Wv, row-reduce
  const float* pmb = pm + (size_t)b * 512 * 128;
  for (int i = 0; i < 32; ++i) {
    int tile = w * 32 + i;
    int mt = (tile >> 3) * 16, nt = (tile & 7) * 16;
    bf16x16 a  = load_frag_A(&s_loc[0][0], 32, mt, 0);
    bf16x16 bb = load_frag_L(WlocT, 32, nt, 0);
    f32x8 acc = {};
    acc = wmma_bf16(a, bb, acc);
    int acol = nt + nl;
    float qv = s_q[acol], wvv = Wv[acol];
    for (int r = 0; r < 8; ++r) {
      int trow = mt + r + half * 8;
      float v = tanhf(acc[r] + qv + pmb[(size_t)trow * 128 + acol]) * wvv;
      for (int off = 8; off > 0; off >>= 1) v += __shfl_xor(v, off, 32);
      if (nl == 0) atomicAdd(&s_e[trow], v);
    }
  }
  __syncthreads();

  int Lb = tlen[b];
  float mx = -3.0e38f;
  for (int i = tid; i < 512; i += 256) {
    float e = (i < Lb) ? s_e[i] : -1.0e9f;
    s_e[i] = e;
    mx = fmaxf(mx, e);
  }
  for (int off = 16; off > 0; off >>= 1) mx = fmaxf(mx, __shfl_xor(mx, off, 32));
  if (lane == 0) s_red[w] = mx;
  __syncthreads();
  if (tid == 0) {
    float m2 = s_red[0];
    for (int j = 1; j < 8; ++j) m2 = fmaxf(m2, s_red[j]);
    s_red[0] = m2;
  }
  __syncthreads();
  mx = s_red[0];
  float sm = 0.0f;
  for (int i = tid; i < 512; i += 256) {
    float e = __expf(s_e[i] - mx);
    s_e[i] = e;
    sm += e;
  }
  for (int off = 16; off > 0; off >>= 1) sm += __shfl_xor(sm, off, 32);
  __syncthreads();
  if (lane == 0) s_red[w] = sm;
  __syncthreads();
  if (tid == 0) {
    float t2 = 0.0f;
    for (int j = 0; j < 8; ++j) t2 += s_red[j];
    s_red[0] = t2;
  }
  __syncthreads();
  float inv = 1.0f / s_red[0];

  for (int i = tid; i < 512; i += 256) {
    float an = s_e[i] * inv;
    s_aw[i] = an;
    aw[b * 512 + i] = an;
    awc[b * 512 + i] = s_awc[i] + an;
    align_out[(size_t)b * 800 * 512 + (size_t)t * 512 + i] = an;
  }
  __syncthreads();

  const float* encb = enc + (size_t)b * 512 * 512;
  for (int e = tid; e < 512; e += 256) {       // context = aw_new @ enc[b]
    float c = 0.0f;
    for (int tt = 0; tt < 512; ++tt) c += s_aw[tt] * encb[(size_t)tt * 512 + e];
    ctxB[b * 512 + e] = (bf16_t)c;
  }
}

// ---------------------------------------------------------------------------
// Projection + gate (gate folded in as column 80). Two clean K-loops.
// ---------------------------------------------------------------------------
__global__ __launch_bounds__(256) void proj_kernel(
    const bf16_t* dhB, const bf16_t* ctxB, const bf16_t* WT,
    const float* bproj, const float* bgate, float* out, int t)
{
  int w = threadIdx.x >> 5, lane = threadIdx.x & 31;
  int half = lane >> 4, nl = lane & 15;
  f32x8 acc[3] = {};
  for (int k = 0; k < 1024; k += 32) {
    for (int i = 0; i < 3; ++i) {
      int id = w * 3 + i;
      int mt = (id / 6) * 16, nt = (id % 6) * 16;
      bf16x16 a = load_frag_A(dhB, 1024, mt, k);
      bf16x16 bb = load_frag_B(WT, 1536, nt, k);
      acc[i] = wmma_bf16(a, bb, acc[i]);
    }
  }
  for (int k = 0; k < 512; k += 32) {
    for (int i = 0; i < 3; ++i) {
      int id = w * 3 + i;
      int mt = (id / 6) * 16, nt = (id % 6) * 16;
      bf16x16 a = load_frag_A(ctxB, 512, mt, k);
      bf16x16 bb = load_frag_B(WT, 1536, nt, 1024 + k);
      acc[i] = wmma_bf16(a, bb, acc[i]);
    }
  }
  for (int i = 0; i < 3; ++i) {
    int id = w * 3 + i;
    int mt = (id / 6) * 16, nt = (id % 6) * 16;
    for (int r = 0; r < 8; ++r) {
      int bm = mt + r + half * 8;
      int n = nt + nl;
      float v = acc[i][r];
      if (n < 80)
        out[(size_t)bm * 80 * 800 + (size_t)n * 800 + t] = v + bproj[n];
      else if (n == 80)
        out[(size_t)4096000 + (size_t)bm * 800 + t] = v + bgate[0];
    }
  }
}

// ---------------------------------------------------------------------------
// Prep kernels
// ---------------------------------------------------------------------------
__global__ void transpose_to_bf16(const float* in, bf16_t* out,
                                  int K, int N, int KP, int Npad, int ldo, int koff)
{
  for (size_t idx = (size_t)blockIdx.x * 256 + threadIdx.x;
       idx < (size_t)Npad * KP; idx += (size_t)gridDim.x * 256) {
    int n = (int)(idx / KP), k = (int)(idx % KP);
    float v = (k < K && n < N) ? in[(size_t)k * N + n] : 0.0f;
    out[(size_t)n * ldo + koff + k] = (bf16_t)v;
  }
}

__global__ void f32_to_bf16(const float* in, bf16_t* out, size_t total) {
  for (size_t i = (size_t)blockIdx.x * 256 + threadIdx.x; i < total;
       i += (size_t)gridDim.x * 256)
    out[i] = (bf16_t)in[i];
}

__global__ void build_xin(const float* dec, bf16_t* out) {
  const size_t total = (size_t)800 * 64 * 96;
  for (size_t idx = (size_t)blockIdx.x * 256 + threadIdx.x; idx < total;
       idx += (size_t)gridDim.x * 256) {
    int k = (int)(idx % 96);
    size_t rest = idx / 96;
    int b = (int)(rest % 64);
    int t = (int)(rest / 64);
    float v = (t > 0 && k < 80) ? dec[(size_t)b * 80 * 800 + (size_t)k * 800 + (t - 1)]
                                : 0.0f;
    out[idx] = (bf16_t)v;
  }
}

__global__ void fill_gate_row(const float* Wgate, bf16_t* WprojT) {
  int k = blockIdx.x * 256 + threadIdx.x;
  if (k < 1536) WprojT[(size_t)80 * 1536 + k] = (bf16_t)Wgate[k];
}

__global__ void zero_u32(unsigned* p, size_t n) {
  for (size_t i = (size_t)blockIdx.x * 256 + threadIdx.x; i < n;
       i += (size_t)gridDim.x * 256)
    p[i] = 0u;
}

// ---------------------------------------------------------------------------
// Host driver
// ---------------------------------------------------------------------------
extern "C" void kernel_launch(void* const* d_in, const int* in_sizes, int n_in,
                              void* d_out, int out_size, void* d_ws, size_t ws_size,
                              hipStream_t stream) {
  (void)in_sizes; (void)n_in; (void)out_size; (void)ws_size;

  const float* enc   = (const float*)d_in[0];
  const float* dec   = (const float*)d_in[1];
  const int*   tlen  = (const int*)  d_in[2];
  const float* Wp1   = (const float*)d_in[3];
  const float* Wp2   = (const float*)d_in[4];
  const float* Wi_a  = (const float*)d_in[5];
  const float* Wh_a  = (const float*)d_in[6];
  const float* b_a   = (const float*)d_in[7];
  const float* Wq    = (const float*)d_in[8];
  const float* Wm    = (const float*)d_in[9];
  const float* Wv    = (const float*)d_in[10];
  const float* Kloc  = (const float*)d_in[11];
  const float* Wloc  = (const float*)d_in[12];
  const float* Wi_d  = (const float*)d_in[13];
  const float* Wh_d  = (const float*)d_in[14];
  const float* b_d   = (const float*)d_in[15];
  const float* Wproj = (const float*)d_in[16];
  const float* bproj = (const float*)d_in[17];
  const float* Wgate = (const float*)d_in[18];
  const float* bgate = (const float*)d_in[19];
  float* out = (float*)d_out;

  char* ws = (char*)d_ws;
  size_t off = 0;
  auto alloc = [&](size_t bytes) -> char* {
    off = (off + 255) & ~(size_t)255;
    char* p = ws + off; off += bytes; return p;
  };

  bf16_t* WaT    = (bf16_t*)alloc((size_t)4096 * 1792 * 2);
  bf16_t* WdT    = (bf16_t*)alloc((size_t)4096 * 2560 * 2);
  bf16_t* WmT    = (bf16_t*)alloc((size_t)128 * 512 * 2);
  bf16_t* WqT    = (bf16_t*)alloc((size_t)128 * 1024 * 2);
  bf16_t* WlocT  = (bf16_t*)alloc((size_t)128 * 32 * 2);
  bf16_t* WprT   = (bf16_t*)alloc((size_t)96 * 1536 * 2);
  bf16_t* Wp1T   = (bf16_t*)alloc((size_t)256 * 96 * 2);
  bf16_t* Wp2T   = (bf16_t*)alloc((size_t)256 * 256 * 2);
  bf16_t* encB   = (bf16_t*)alloc((size_t)64 * 512 * 512 * 2);
  float*  pmBuf  = (float*) alloc((size_t)64 * 512 * 128 * 4);
  bf16_t* xin    = (bf16_t*)alloc((size_t)800 * 64 * 96 * 2);
  bf16_t* h1     = (bf16_t*)alloc((size_t)800 * 64 * 256 * 2);
  bf16_t* hpre   = (bf16_t*)alloc((size_t)800 * 64 * 256 * 2);

  size_t stateOff = (off + 255) & ~(size_t)255;
  float*  ahF   = (float*) alloc((size_t)64 * 1024 * 4);
  float*  acF   = (float*) alloc((size_t)64 * 1024 * 4);
  float*  dhF   = (float*) alloc((size_t)64 * 1024 * 4);
  float*  dcF   = (float*) alloc((size_t)64 * 1024 * 4);
  float*  awB   = (float*) alloc((size_t)64 * 512 * 4);
  float*  awcB  = (float*) alloc((size_t)64 * 512 * 4);
  bf16_t* ahB0  = (bf16_t*)alloc((size_t)64 * 1024 * 2);
  bf16_t* ahB1  = (bf16_t*)alloc((size_t)64 * 1024 * 2);
  bf16_t* dhB0  = (bf16_t*)alloc((size_t)64 * 1024 * 2);
  bf16_t* dhB1  = (bf16_t*)alloc((size_t)64 * 1024 * 2);
  bf16_t* ctxB  = (bf16_t*)alloc((size_t)64 * 512 * 2);
  size_t stateBytes = off - stateOff;

  dim3 blk(256);
  auto TGRID = [&](size_t total) { size_t g = (total + 255) / 256;
                                   return (unsigned)(g > 16384 ? 16384 : g); };

  transpose_to_bf16<<<TGRID((size_t)4096*768),  blk, 0, stream>>>(Wi_a,  WaT, 768, 4096, 768, 4096, 1792, 0);
  transpose_to_bf16<<<TGRID((size_t)4096*1024), blk, 0, stream>>>(Wh_a,  WaT, 1024,4096,1024, 4096, 1792, 768);
  transpose_to_bf16<<<TGRID((size_t)4096*1536), blk, 0, stream>>>(Wi_d,  WdT, 1536,4096,1536, 4096, 2560, 0);
  transpose_to_bf16<<<TGRID((size_t)4096*1024), blk, 0, stream>>>(Wh_d,  WdT, 1024,4096,1024, 4096, 2560, 1536);
  transpose_to_bf16<<<TGRID((size_t)128*512),   blk, 0, stream>>>(Wm,    WmT, 512, 128, 512, 128, 512, 0);
  transpose_to_bf16<<<TGRID((size_t)128*1024),  blk, 0, stream>>>(Wq,    WqT, 1024,128, 1024,128, 1024, 0);
  transpose_to_bf16<<<TGRID((size_t)128*32),    blk, 0, stream>>>(Wloc,  WlocT,32, 128, 32,  128, 32,  0);
  transpose_to_bf16<<<TGRID((size_t)96*1536),   blk, 0, stream>>>(Wproj, WprT, 1536,80, 1536,96, 1536, 0);
  fill_gate_row<<<6, blk, 0, stream>>>(Wgate, WprT);
  transpose_to_bf16<<<TGRID((size_t)256*96),    blk, 0, stream>>>(Wp1,   Wp1T, 80, 256, 96, 256, 96,  0);
  transpose_to_bf16<<<TGRID((size_t)256*256),   blk, 0, stream>>>(Wp2,   Wp2T, 256,256, 256,256, 256, 0);
  f32_to_bf16<<<8192, blk, 0, stream>>>(enc, encB, (size_t)64 * 512 * 512);

  build_xin<<<8192, blk, 0, stream>>>(dec, xin);
  gemm1_kernel<<<dim3(4, 800), blk, 0, stream>>>(xin, Wp1T, 96, 96, nullptr, h1, 256, 1);
  gemm1_kernel<<<dim3(4, 800), blk, 0, stream>>>(h1,  Wp2T, 256, 256, nullptr, hpre, 256, 1);
  gemm1_kernel<<<dim3(2, 512), blk, 0, stream>>>(encB, WmT, 512, 512, pmBuf, nullptr, 128, 0);

  zero_u32<<<1024, blk, 0, stream>>>((unsigned*)(ws + stateOff), stateBytes / 4);

  float* alignOut = out + (size_t)4096000 + 51200;

  for (int t = 0; t < 800; ++t) {
    int par = t & 1;
    bf16_t* ahIn  = par ? ahB1 : ahB0;
    bf16_t* ahOut = par ? ahB0 : ahB1;
    bf16_t* dhIn  = par ? dhB1 : dhB0;
    bf16_t* dhOut = par ? dhB0 : dhB1;

    lstm_kernel<<<64, blk, 0, stream>>>(
        hpre + (size_t)t * 64 * 256, 256, ctxB, 512, ahIn, 1024,
        WaT, 1792, 1792, b_a, acF, ahF, ahOut);

    attention_kernel<<<64, blk, 0, stream>>>(
        ahF, enc, pmBuf, WqT, WlocT, Kloc, Wv, tlen,
        awB, awcB, ctxB, alignOut, t);

    lstm_kernel<<<64, blk, 0, stream>>>(
        ahOut, 1024, ctxB, 512, dhIn, 1024,
        WdT, 2560, 2560, b_d, dcF, dhF, dhOut);

    proj_kernel<<<1, blk, 0, stream>>>(dhOut, ctxB, WprT, bproj, bgate, out, t);
  }
}
